// Attention_24927990186215
// MI455X (gfx1250) — compile-verified, matrix operand order checked by reading, not run on hardware
//
#include <hip/hip_runtime.h>

typedef __bf16 bf16_t;
typedef __attribute__((ext_vector_type(16))) __bf16 v16bf;
typedef __attribute__((ext_vector_type(8)))  __bf16 v8bf;
typedef __attribute__((ext_vector_type(4)))  __bf16 v4bf;
typedef __attribute__((ext_vector_type(8)))  float  v8f;

#define DD  512
#define TT  16
#define NH  8
#define HS  64
#define WPB 2              // windows per block
#define WSZ (TT * DD)      // 8192 elements per window tile
// packed weight block: [nt(32)][kt(16)][lane(32)][elem(16)] bf16, per 512x512 matrix

// ---------------- weight pack: f32 [K][N] -> bf16 fragment-linear ----------------
__global__ __launch_bounds__(256)
void pack_weights(const float* __restrict__ Wq, const float* __restrict__ Wk,
                  const float* __restrict__ Wv, const float* __restrict__ Wo,
                  bf16_t* __restrict__ dst) {
  int wsel = blockIdx.y;
  const float* W = (wsel == 0) ? Wq : (wsel == 1) ? Wk : (wsel == 2) ? Wv : Wo;
  int p = blockIdx.x * blockDim.x + threadIdx.x;      // 0 .. 512*512-1
  int elem = p & 15;
  int lane = (p >> 4) & 31;
  int kt   = (p >> 9) & 15;
  int nt   = p >> 13;
  int n  = nt * 16 + (lane & 15);
  int hf = lane >> 4;
  int kl = (elem < 8) ? (hf * 8 + elem) : (16 + hf * 8 + (elem - 8));
  int k  = kt * 32 + kl;
  dst[(size_t)wsel * (DD * DD) + p] = (bf16_t)W[k * DD + n];
}

// ---------------- fragment helpers ----------------
// A-fragment (16x32 bf16) from row-major LDS, row stride `stride` elements.
__device__ __forceinline__ v16bf frag_a_lds(const bf16_t* base, int stride, int lane) {
  int m  = lane & 15;
  int hf = lane >> 4;
  const bf16_t* p = base + m * stride + hf * 8;
  v8bf lo = *(const v8bf*)(p);
  v8bf hi = *(const v8bf*)(p + 16);
  v16bf r;
#pragma unroll
  for (int i = 0; i < 8; ++i) { r[i] = lo[i]; r[i + 8] = hi[i]; }
  return r;
}

// B-fragment from packed weights: 32 bytes contiguous per lane.
__device__ __forceinline__ v16bf frag_b_w(const bf16_t* tile, int lane) {
  const bf16_t* p = tile + lane * 16;
  v8bf lo = *(const v8bf*)(p);
  v8bf hi = *(const v8bf*)(p + 8);
  v16bf r;
#pragma unroll
  for (int i = 0; i < 8; ++i) { r[i] = lo[i]; r[i + 8] = hi[i]; }
  return r;
}

__device__ __forceinline__ v8f wmma_bf16(v16bf a, v16bf b, v8f c) {
  return __builtin_amdgcn_wmma_f32_16x16x32_bf16(false, a, false, b, (short)0, c,
                                                 false, false);
}

// ---------------- fused windowed attention, 2 windows / block ----------------
__global__ __launch_bounds__(256)
void win_attn(const float* __restrict__ x,
              const float* __restrict__ bq, const float* __restrict__ bk,
              const float* __restrict__ bv, const float* __restrict__ bo,
              const bf16_t* __restrict__ Wpk,   // 4 packed 512x512 bf16 matrices
              float* __restrict__ out) {
  extern __shared__ __align__(16) char smem_raw[];
  bf16_t* q_s = (bf16_t*)smem_raw;          // [WPB][8192] 32KB; reused as ctx
  bf16_t* k_s = q_s + WPB * WSZ;            // [WPB][8192] 32KB
  bf16_t* v_s = k_s + WPB * WSZ;            // [WPB][8192] 32KB
  bf16_t* xb  = v_s + WPB * WSZ;            // [WPB][8192] 32KB; then probs staging 8KB
  bf16_t* stag = xb;                        // [WPB][NH][16][16] bf16 probs

  const int tid  = threadIdx.x;
  const int lane = tid & 31;
  const int wave = tid >> 5;
  const int hf   = lane >> 4;
  const int nn   = lane & 15;
  const size_t win0 = (size_t)blockIdx.x * WPB;

  const float* xw = x + win0 * WSZ;         // 2 consecutive windows are contiguous

  // ---- phase 0: load x for both windows, convert to bf16 in LDS ----
#pragma unroll
  for (int j = 0; j < 8 * WPB; ++j) {
    int idx = tid + 256 * j;                // float4 index, 4096 total
    float4 f = ((const float4*)xw)[idx];
    v4bf b;
    b[0] = (bf16_t)f.x; b[1] = (bf16_t)f.y; b[2] = (bf16_t)f.z; b[3] = (bf16_t)f.w;
    *((v4bf*)&xb[idx * 4]) = b;
  }
  __syncthreads();

  // ---- phase 1: q,k,v = x @ W{q,k,v} + b  (96 tiles/window; B reused 2x) ----
#pragma unroll 1
  for (int tI = 0; tI < 12; ++tI) {
    int tau   = wave + 8 * tI;              // 0..95
    int which = tau >> 5;                   // 0=q 1=k 2=v
    int nt    = tau & 31;
    const bf16_t* wtile = Wpk + (size_t)which * (DD * DD) + (size_t)nt * 8192;
    v8f acc0 = {}, acc1 = {};
#pragma unroll
    for (int kt = 0; kt < 16; ++kt) {
      v16bf b  = frag_b_w(wtile + kt * 512, lane);
      v16bf a0 = frag_a_lds(xb + 0 * WSZ + kt * 32, DD, lane);
      v16bf a1 = frag_a_lds(xb + 1 * WSZ + kt * 32, DD, lane);
      acc0 = wmma_bf16(a0, b, acc0);
      acc1 = wmma_bf16(a1, b, acc1);
    }
    const float* bp = (which == 0) ? bq : (which == 1) ? bk : bv;
    float bias = bp[nt * 16 + nn];
    bf16_t* dst = ((which == 0) ? q_s : (which == 1) ? k_s : v_s) + nt * 16;
#pragma unroll
    for (int r = 0; r < 8; ++r) {
      dst[(r + hf * 8) * DD + nn]       = (bf16_t)(acc0[r] + bias);
      dst[WSZ + (r + hf * 8) * DD + nn] = (bf16_t)(acc1[r] + bias);
    }
  }
  __syncthreads();

  // ---- phase 2a: scores = sigmoid(q_h k_h^T / 8); wave = head, loop windows ----
#pragma unroll
  for (int w = 0; w < WPB; ++w) {
    int h = wave;
    v8f sc = {};
#pragma unroll
    for (int kt = 0; kt < 2; ++kt) {
      v16bf a = frag_a_lds(q_s + w * WSZ + h * HS + kt * 32, DD, lane);
      // B-fragment of k^T == A-style row load of k
      v16bf b = frag_a_lds(k_s + w * WSZ + h * HS + kt * 32, DD, lane);
      sc = wmma_bf16(a, b, sc);
    }
    bf16_t* st = stag + (w * NH + h) * 256;
#pragma unroll
    for (int r = 0; r < 8; ++r) {
      float pr = 1.0f / (1.0f + __expf(-sc[r] * 0.125f));
      st[(r + hf * 8) * 16 + nn] = (bf16_t)pr;
    }
  }
  __syncthreads();   // probs visible; all q reads done -> q_s reusable as ctx

  // ---- phase 2b: ctx_h = probs_h @ v_h  (K=16 zero-padded to 32) ----
#pragma unroll
  for (int w = 0; w < WPB; ++w) {
    int h = wave;
    v16bf a;
    {
      const bf16_t* p = stag + (w * NH + h) * 256 + nn * 16 + hf * 8;
      v8bf lo = *(const v8bf*)p;
#pragma unroll
      for (int i = 0; i < 8; ++i) { a[i] = lo[i]; a[i + 8] = (bf16_t)0.0f; }
    }
#pragma unroll
    for (int dt = 0; dt < 4; ++dt) {
      const bf16_t* vb = v_s + w * WSZ + h * HS + dt * 16;
      v16bf b;
#pragma unroll
      for (int i = 0; i < 8; ++i) {          // duplicate halves: padded K hits A==0
        bf16_t e = vb[(hf * 8 + i) * DD + nn];
        b[i] = e; b[i + 8] = e;
      }
      v8f acc = {};
      acc = wmma_bf16(a, b, acc);
      bf16_t* dst = q_s + w * WSZ + h * HS + dt * 16;   // ctx overwrites q region
#pragma unroll
      for (int r = 0; r < 8; ++r) dst[(r + hf * 8) * DD + nn] = (bf16_t)acc[r];
    }
  }
  __syncthreads();

  // ---- phase 3: out = ctx @ Wo + bo  (B reused 2x) ----
  const bf16_t* wo = Wpk + (size_t)3 * (DD * DD);
  float* ow = out + win0 * WSZ;
#pragma unroll 1
  for (int tI = 0; tI < 4; ++tI) {
    int nt = wave + 8 * tI;                  // 0..31
    const bf16_t* wtile = wo + (size_t)nt * 8192;
    v8f acc0 = {}, acc1 = {};
#pragma unroll
    for (int kt = 0; kt < 16; ++kt) {
      v16bf b  = frag_b_w(wtile + kt * 512, lane);
      v16bf a0 = frag_a_lds(q_s + 0 * WSZ + kt * 32, DD, lane);
      v16bf a1 = frag_a_lds(q_s + 1 * WSZ + kt * 32, DD, lane);
      acc0 = wmma_bf16(a0, b, acc0);
      acc1 = wmma_bf16(a1, b, acc1);
    }
    float bias = bo[nt * 16 + nn];
#pragma unroll
    for (int r = 0; r < 8; ++r) {
      ow[(size_t)(r + hf * 8) * DD + nt * 16 + nn]       = acc0[r] + bias;
      ow[WSZ + (size_t)(r + hf * 8) * DD + nt * 16 + nn] = acc1[r] + bias;
    }
  }
}

extern "C" void kernel_launch(void* const* d_in, const int* in_sizes, int n_in,
                              void* d_out, int out_size, void* d_ws, size_t ws_size,
                              hipStream_t stream) {
  const float* x  = (const float*)d_in[0];
  const float* Wq = (const float*)d_in[1];
  const float* bq = (const float*)d_in[2];
  const float* Wk = (const float*)d_in[3];
  const float* bk = (const float*)d_in[4];
  const float* Wv = (const float*)d_in[5];
  const float* bv = (const float*)d_in[6];
  const float* Wo = (const float*)d_in[7];
  const float* bo = (const float*)d_in[8];
  bf16_t* Wpk = (bf16_t*)d_ws;               // 4 * 512*512 * 2B = 2 MB

  dim3 gp(DD * DD / 256, 4);
  pack_weights<<<gp, 256, 0, stream>>>(Wq, Wk, Wv, Wo, Wpk);

  int nwin = 16 * 16 * 16;                   // 4096 windows
  size_t smem = (size_t)4 * WPB * WSZ * sizeof(bf16_t);   // 128 KB
  win_attn<<<nwin / WPB, 256, smem, stream>>>(x, bq, bk, bv, bo, Wpk, (float*)d_out);
}
